// Classify_Anything_Mixed_OT_Loss_43490838839624
// MI455X (gfx1250) — matrix operand order repaired and checked by reading, not compiled, not run on hardware
//
#include <hip/hip_runtime.h>
#include <hip/hip_bf16.h>

#define B_DIM 4096
#define C_DIM 8192
#define D_DIM 512
#define TEMP_INV (1.0f / 0.07f)

typedef __attribute__((ext_vector_type(16))) __bf16 bf16x16;
typedef __attribute__((ext_vector_type(8)))  float  f32x8;

// ---------------------------------------------------------------------------
// GEMM: sim[B,C] = (features @ text^T) / TEMP via bf16 WMMA with hi/lo split
// (3 bf16 products recover ~fp32 precision; memory-bound overall so GEMM
//  must run at matrix-core rates, not the slow f32 K=4 path).
// 128x128 tile per block, 256 threads = 8 waves, each wave owns 32x64.
// ---------------------------------------------------------------------------
#define TM 128
#define TN 128
#define TK 32
#define LDSTR 34   // 32 + 2 bf16 pad -> 17 dwords/row, conflict-free

__device__ __forceinline__ void cvt_store4(__bf16* hi, __bf16* lo, int base, float4 v) {
    float vals[4] = {v.x, v.y, v.z, v.w};
#pragma unroll
    for (int j = 0; j < 4; ++j) {
        __bf16 h = (__bf16)vals[j];
        float hf = (float)h;
        hi[base + j] = h;
        lo[base + j] = (__bf16)(vals[j] - hf);
    }
}

__global__ __launch_bounds__(256) void gemm_sim(const float* __restrict__ A,
                                                const float* __restrict__ Bm,
                                                float* __restrict__ sim) {
    __shared__ __bf16 Ahi[TM * LDSTR], Alo[TM * LDSTR];
    __shared__ __bf16 Bhi[TN * LDSTR], Blo[TN * LDSTR];

    const int tid  = threadIdx.x;
    const int lane = tid & 31;
    const int wave = tid >> 5;
    const int waveM = wave & 3;   // rows [waveM*32, +32)
    const int waveN = wave >> 2;  // cols [waveN*64, +64)
    const int rowBase = blockIdx.y * TM;
    const int colBase = blockIdx.x * TN;

    const int mrow   = lane & 15;
    const int laneHi = lane >> 4;
    const int kgrpA  = laneHi ? 8 : 0;   // unused: A fragments also use kgrpB (K=16..31 in lanes16-31)
    const int kgrpB  = laneHi ? 16 : 0;  // B layout: lanes16-31 hold K=16..31

    f32x8 acc[2][4];
#pragma unroll
    for (int mt = 0; mt < 2; ++mt)
#pragma unroll
        for (int nt = 0; nt < 4; ++nt) acc[mt][nt] = (f32x8){0,0,0,0,0,0,0,0};

    for (int k0 = 0; k0 < D_DIM; k0 += TK) {
        __syncthreads();
        // Stage 128x32 f32 tiles of A and B, split into bf16 hi/lo, into LDS.
#pragma unroll
        for (int i = 0; i < 4; ++i) {
            int idx = tid + i * 256;       // float4 unit index, 0..1023
            int r   = idx >> 3;            // 8 float4 per row
            int c4  = idx & 7;
            int base = r * LDSTR + c4 * 4;
            float4 av = *(const float4*)(A  + (size_t)(rowBase + r) * D_DIM + k0 + c4 * 4);
            cvt_store4(Ahi, Alo, base, av);
            float4 bv = *(const float4*)(Bm + (size_t)(colBase + r) * D_DIM + k0 + c4 * 4);
            cvt_store4(Bhi, Blo, base, bv);
        }
        __syncthreads();

        // Build A fragments (16x32, contiguous K: lanes16-31 hold K=16..31, same order as B)
        bf16x16 afh[2], afl[2];
#pragma unroll
        for (int mt = 0; mt < 2; ++mt) {
            int M = waveM * 32 + mt * 16 + mrow;
            const __bf16* ph = &Ahi[M * LDSTR];
            const __bf16* pl = &Alo[M * LDSTR];
#pragma unroll
            for (int e = 0; e < 16; ++e) {
                int K = kgrpB + e;
                afh[mt][e] = ph[K];
                afl[mt][e] = pl[K];
            }
        }
        // Build B fragments (16-bit B 32x16 layout)
        bf16x16 bfh[4], bfl[4];
#pragma unroll
        for (int nt = 0; nt < 4; ++nt) {
            int N = waveN * 64 + nt * 16 + mrow;
            const __bf16* ph = &Bhi[N * LDSTR];
            const __bf16* pl = &Blo[N * LDSTR];
#pragma unroll
            for (int e = 0; e < 16; ++e) {
                int K = kgrpB + e;
                bfh[nt][e] = ph[K];
                bfl[nt][e] = pl[K];
            }
        }
        // 3-product compensated bf16 GEMM: hi*hi + hi*lo + lo*hi
#pragma unroll
        for (int mt = 0; mt < 2; ++mt)
#pragma unroll
            for (int nt = 0; nt < 4; ++nt) {
                acc[mt][nt] = __builtin_amdgcn_wmma_f32_16x16x32_bf16(
                    false, afh[mt], false, bfh[nt], (short)0, acc[mt][nt], false, false);
                acc[mt][nt] = __builtin_amdgcn_wmma_f32_16x16x32_bf16(
                    false, afh[mt], false, bfl[nt], (short)0, acc[mt][nt], false, false);
                acc[mt][nt] = __builtin_amdgcn_wmma_f32_16x16x32_bf16(
                    false, afl[mt], false, bfh[nt], (short)0, acc[mt][nt], false, false);
            }
    }

    // Store: C/D layout -> VGPR v: M = v + (laneHi?8:0), N = lane&15
#pragma unroll
    for (int mt = 0; mt < 2; ++mt) {
        int rowStart = rowBase + waveM * 32 + mt * 16 + laneHi * 8;
#pragma unroll
        for (int nt = 0; nt < 4; ++nt) {
            int col = colBase + waveN * 64 + nt * 16 + mrow;
#pragma unroll
            for (int v = 0; v < 8; ++v) {
                sim[(size_t)(rowStart + v) * C_DIM + col] = acc[mt][nt][v] * TEMP_INV;
            }
        }
    }
}

// ---------------------------------------------------------------------------
// Per-row: rowsum0 = sum_c exp(sim), diag = sim[i,i], R = 1
// ---------------------------------------------------------------------------
__global__ __launch_bounds__(256) void row_stats(const float* __restrict__ sim,
                                                 float* rowsum0, float* Rv, float* diag) {
    __shared__ float red[256];
    int row = blockIdx.x, tid = threadIdx.x;
    const float* p = sim + (size_t)row * C_DIM;
    float s = 0.f;
    for (int c = tid; c < C_DIM; c += 256) s += __expf(p[c]);
    red[tid] = s; __syncthreads();
    for (int st = 128; st; st >>= 1) { if (tid < st) red[tid] += red[tid + st]; __syncthreads(); }
    if (tid == 0) { rowsum0[row] = red[0]; Rv[row] = 1.0f; diag[row] = p[row]; }
}

// m = count(idx==1) + 0.1*count(idx==0); idx in {0,1}
__global__ __launch_bounds__(256) void compute_m(const int* __restrict__ dsidx, float* mval) {
    __shared__ float red[256];
    int tid = threadIdx.x;
    float acc = 0.f;
    for (int i = tid; i < B_DIM; i += 256) acc += (dsidx[i] == 1) ? 1.0f : 0.1f;
    red[tid] = acc; __syncthreads();
    for (int st = 128; st; st >>= 1) { if (tid < st) red[tid] += red[tid + st]; __syncthreads(); }
    if (tid == 0) mval[0] = red[0];
}

// colsum partial: part[by][c] = sum_{r in chunk} exp(sim[r,c]) * R[r], c < B only
__global__ __launch_bounds__(256) void colsum_partial(const float* __restrict__ sim,
                                                      const float* __restrict__ Rv,
                                                      float* __restrict__ part) {
    int c  = blockIdx.x * 256 + threadIdx.x;   // 0..4095
    int rb = blockIdx.y * 256;
    float acc = 0.f;
    for (int r = rb; r < rb + 256; ++r)
        acc += __expf(sim[(size_t)r * C_DIM + c]) * Rv[r];
    part[(size_t)blockIdx.y * B_DIM + c] = acc;
}

// d[i] = min(diag[i]/colsum[i], 1); blockS partials of sum d*R*rowsum0
__global__ __launch_bounds__(256) void make_dS(const float* __restrict__ part,
                                               const float* __restrict__ diag,
                                               const float* __restrict__ Rv,
                                               const float* __restrict__ rowsum0,
                                               float* dvec, float* blockS) {
    __shared__ float red[256];
    int tid = threadIdx.x;
    int i = blockIdx.x * 256 + tid;
    float cs = 0.f;
#pragma unroll
    for (int j = 0; j < 16; ++j) cs += part[(size_t)j * B_DIM + i];
    float d = fminf(diag[i] / cs, 1.0f);
    dvec[i] = d;
    red[tid] = d * Rv[i] * rowsum0[i]; __syncthreads();
    for (int st = 128; st; st >>= 1) { if (tid < st) red[tid] += red[tid + st]; __syncthreads(); }
    if (tid == 0) blockS[blockIdx.x] = red[0];
}

__global__ __launch_bounds__(256) void update_R(const float* __restrict__ blockS,
                                                const float* __restrict__ dvec,
                                                const float* __restrict__ mval,
                                                float* __restrict__ Rv) {
    float S = 0.f;
#pragma unroll
    for (int j = 0; j < 16; ++j) S += blockS[j];
    float g = mval[0] / S;
    int i = blockIdx.x * 256 + threadIdx.x;
    Rv[i] *= dvec[i] * g;
}

// ---------------------------------------------------------------------------
// Final fused pass: per-row log-softmax of P = R*exp(sim), targets dot,
// ASL elementwise, masked coco/cifar copies. One block per row; row staged
// in 32KB LDS (320KB/WGP makes this cheap).
// ---------------------------------------------------------------------------
__device__ __forceinline__ float blk_sum(float v, float* red, int tid) {
    __syncthreads();
    red[tid] = v; __syncthreads();
    for (int st = 128; st; st >>= 1) { if (tid < st) red[tid] += red[tid + st]; __syncthreads(); }
    return red[0];
}

__global__ __launch_bounds__(256) void final_pass(const float* __restrict__ sim,
                                                  const float* __restrict__ targets,
                                                  const int* __restrict__ dsidx,
                                                  const float* __restrict__ Rv,
                                                  float* __restrict__ mcrow,
                                                  float* __restrict__ aslrow,
                                                  float* __restrict__ coco_out,
                                                  float* __restrict__ cifar_out) {
    __shared__ float rowbuf[C_DIM];   // 32KB
    __shared__ float red[256];
    int row = blockIdx.x, tid = threadIdx.x;
    const float* ps = sim + (size_t)row * C_DIM;
    const float* pt = targets + (size_t)row * C_DIM;
    float R = Rv[row];
    int ds = dsidx[row];
    float cocoMask = (ds == 0) ? 1.0f : 0.0f;
    float cifMask  = (ds == 1) ? 1.0f : 0.0f;

    float mx = -3.402823466e+38f;
    for (int c = tid; c < C_DIM; c += 256) {
        float s = ps[c];
        rowbuf[c] = s;
        mx = fmaxf(mx, s);
    }
    red[tid] = mx; __syncthreads();
    for (int st = 128; st; st >>= 1) { if (tid < st) red[tid] = fmaxf(red[tid], red[tid + st]); __syncthreads(); }
    float smax = red[0];
    float Pmax = R * __expf(smax);    // max of P row (R > 0, exp monotone)

    float se = 0.f, tdot = 0.f, tsum = 0.f, asl = 0.f;
    for (int c = tid; c < C_DIM; c += 256) {
        float s = rowbuf[c];
        float t = pt[c];
        float P = R * __expf(s);
        se   += __expf(P - Pmax);
        tdot += t * P;
        tsum += t;
        // ASL (gamma_neg=4, gamma_pos=1, clip=0.05)
        float xp = 1.0f / (1.0f + __expf(-s));
        float xn = fminf(1.0f - xp + 0.05f, 1.0f);
        float L  = t * __logf(fmaxf(xp, 1e-8f)) + (1.0f - t) * __logf(fmaxf(xn, 1e-8f));
        float ptm = xp * t + xn * (1.0f - t);
        float gamma = t + 4.0f * (1.0f - t);
        L *= __powf(1.0f - ptm, gamma);
        asl += L;
        size_t o = (size_t)row * C_DIM + c;
        coco_out[o]  = cocoMask * s;
        cifar_out[o] = cifMask * s;
    }
    se   = blk_sum(se,   red, tid);
    tdot = blk_sum(tdot, red, tid);
    tsum = blk_sum(tsum, red, tid);
    asl  = blk_sum(asl,  red, tid);
    if (tid == 0) {
        float lse = Pmax + __logf(se);
        mcrow[row]  = tdot - tsum * lse;
        aslrow[row] = asl;
    }
}

__global__ __launch_bounds__(256) void reduce_final(const float* __restrict__ mcrow,
                                                    const float* __restrict__ aslrow,
                                                    float* __restrict__ out) {
    __shared__ float r1[256], r2[256];
    int tid = threadIdx.x;
    float s1 = 0.f, s2 = 0.f;
    for (int i = tid; i < B_DIM; i += 256) { s1 += mcrow[i]; s2 += aslrow[i]; }
    r1[tid] = s1; r2[tid] = s2; __syncthreads();
    for (int st = 128; st; st >>= 1) {
        if (tid < st) { r1[tid] += r1[tid + st]; r2[tid] += r2[tid + st]; }
        __syncthreads();
    }
    if (tid == 0) {
        float mc = -r1[0] / (float)B_DIM;
        float ml = -r2[0];
        out[0] = mc + ml;   // total_loss
        out[1] = mc;        // multiclass_loss
        out[2] = ml;        // multilabel_loss
    }
}

// ---------------------------------------------------------------------------
extern "C" void kernel_launch(void* const* d_in, const int* in_sizes, int n_in,
                              void* d_out, int out_size, void* d_ws, size_t ws_size,
                              hipStream_t stream) {
    (void)in_sizes; (void)n_in; (void)out_size; (void)ws_size;
    const float* features = (const float*)d_in[0];
    const float* textf    = (const float*)d_in[1];
    const float* targets  = (const float*)d_in[2];
    const int*   dsidx    = (const int*)d_in[3];
    float* out = (float*)d_out;
    float* ws  = (float*)d_ws;

    const size_t BC = (size_t)B_DIM * C_DIM;
    float* sim     = ws;
    float* diag    = ws + BC;
    float* rowsum0 = diag + B_DIM;
    float* Rv      = rowsum0 + B_DIM;
    float* dvec    = Rv + B_DIM;
    float* part    = dvec + B_DIM;         // 16 * B_DIM
    float* blockS  = part + 16 * B_DIM;    // 16
    float* mval    = blockS + 16;          // 1 (padded)
    float* mcrow   = mval + 16;
    float* aslrow  = mcrow + B_DIM;

    dim3 ggrid(C_DIM / TN, B_DIM / TM);
    gemm_sim<<<ggrid, 256, 0, stream>>>(features, textf, sim);
    row_stats<<<B_DIM, 256, 0, stream>>>(sim, rowsum0, Rv, diag);
    compute_m<<<1, 256, 0, stream>>>(dsidx, mval);
    for (int it = 0; it < 5; ++it) {
        colsum_partial<<<dim3(16, 16), 256, 0, stream>>>(sim, Rv, part);
        make_dS<<<16, 256, 0, stream>>>(part, diag, Rv, rowsum0, dvec, blockS);
        update_R<<<16, 256, 0, stream>>>(blockS, dvec, mval, Rv);
    }
    final_pass<<<B_DIM, 256, 0, stream>>>(sim, targets, dsidx, Rv, mcrow, aslrow,
                                          out + 3, out + 3 + BC);
    reduce_final<<<1, 256, 0, stream>>>(mcrow, aslrow, out);
}
